// CrossModalMemoryCompressor_13932873908479
// MI455X (gfx1250) — compile-verified, matrix operand order checked by reading, not run on hardware
//
#include <hip/hip_runtime.h>

#define BN    16
#define SEQ   2048
#define DIM   1024
#define RNK   64
#define NKEEP 1024

typedef __attribute__((ext_vector_type(16))) __bf16    v16bf;
typedef __attribute__((ext_vector_type(8)))  float     v8f;
typedef __attribute__((ext_vector_type(4)))  unsigned  v4u;
typedef __attribute__((ext_vector_type(2)))  float     v2f;

union FragBF { v16bf v; v4u q[2]; unsigned short h[16]; };

__device__ __forceinline__ unsigned short f2bf(float f) {
  unsigned int x = __float_as_uint(f);
  x += 0x7FFFu + ((x >> 16) & 1u);       // round-to-nearest-even
  return (unsigned short)(x >> 16);
}
__device__ __forceinline__ unsigned pack_bf2(float lo, float hi) {
  return ((unsigned)f2bf(hi) << 16) | (unsigned)f2bf(lo);
}

// ---------------------------------------------------------------------------
// One-shot: convert projection weights [64,1024] f32 -> bf16 (L2-resident)
// ---------------------------------------------------------------------------
__global__ void wconv_kernel(const float* __restrict__ W,
                             unsigned* __restrict__ Wbf) {
  int p = blockIdx.x * blockDim.x + threadIdx.x;     // pair index
  v2f w = ((const v2f*)W)[p];
  Wbf[p] = pack_bf2(w.x, w.y);
}

// ---------------------------------------------------------------------------
// Projection: out[b, s, r] (bf16) = feat[b, s, :] . W[r, :]
// One workgroup = (b, 16-row tile); wave w owns output cols [16w, 16w+16).
// Feature tile converted once to bf16 in LDS; all fragments are 2x b128 loads.
// ---------------------------------------------------------------------------
__global__ void proj_kernel(const float* __restrict__ feat,
                            const unsigned short* __restrict__ Wbf,
                            unsigned short* __restrict__ out) {
  __shared__ unsigned ldsAu[16 * DIM / 2];
  const unsigned short* ldsA = (const unsigned short*)ldsAu;
  const int b    = blockIdx.y;
  const int s0   = blockIdx.x * 16;
  const int tid  = threadIdx.x;
  const int lane = tid & 31;
  const int wave = tid >> 5;

  // rows s0..s0+15 are contiguous in memory -> linear packed staging
  const v2f* fbase = (const v2f*)(feat + ((size_t)b * SEQ + s0) * DIM);
  for (int p = tid; p < 16 * DIM / 2; p += blockDim.x) {
    v2f v = fbase[p];
    ldsAu[p] = pack_bf2(v.x, v.y);
  }
  __syncthreads();

  const int M  = lane & 15;
  const int g0 = (lane < 16) ? 0 : 8;   // ISA 16-bit A/B fragment layout
  const int n  = lane & 15;
  const int r0 = wave * 16;

  v8f acc = {};
  for (int kk = 0; kk < DIM; kk += 32) {
    FragBF a, bm;
    const unsigned short* ar = ldsA + M * DIM + kk + g0;
    const unsigned short* wr = Wbf + (size_t)(r0 + n) * DIM + kk + g0;
    __builtin_prefetch(wr + 64, 0, 1);
    a.q[0]  = *(const v4u*)ar;
    a.q[1]  = *(const v4u*)(ar + 16);
    bm.q[0] = *(const v4u*)wr;
    bm.q[1] = *(const v4u*)(wr + 16);
    acc = __builtin_amdgcn_wmma_f32_16x16x32_bf16(false, a.v, false, bm.v,
                                                  (short)0, acc, false, false);
  }
  const int roff = (lane < 16) ? 0 : 8;  // ISA f32 C/D layout
#pragma unroll
  for (int i = 0; i < 8; ++i) {
    int row = s0 + i + roff;
    out[((size_t)b * SEQ + row) * RNK + r0 + n] = f2bf(acc[i]);
  }
}

// ---------------------------------------------------------------------------
// imp[row] = addc + 0.1 * ||feat[row, :]||_2   (one wave per row)
// vision: addc = 1.0 (softmax row-sum is exactly 1); lang: addc = 0 (base)
// ---------------------------------------------------------------------------
__global__ void norm_imp_kernel(const float* __restrict__ feat,
                                float* __restrict__ imp, float addc) {
  const int lane = threadIdx.x & 31;
  const int wave = threadIdx.x >> 5;
  const int row  = blockIdx.x * (blockDim.x >> 5) + wave;
  const float* f = feat + (size_t)row * DIM;
  float s = 0.f;
  for (int d = lane; d < DIM; d += 32) { float v = f[d]; s += v * v; }
#pragma unroll
  for (int m = 16; m; m >>= 1) s += __shfl_xor(s, m, 32);
  if (lane == 0) imp[row] = addc + 0.1f * sqrtf(s);
}

// ---------------------------------------------------------------------------
// Pass 1: per vision-row softmax stats (max m, denom Z) over all 2048 cols.
// Scores via 2x WMMA bf16 16x16x32 (K = 64); fragments are b128 global loads.
// ---------------------------------------------------------------------------
__global__ void rowstats_kernel(const unsigned short* __restrict__ qv,
                                const unsigned short* __restrict__ kl,
                                float* __restrict__ m_ws,
                                float* __restrict__ z_ws) {
  __shared__ float lm[4][16], lz[4][16];
  const int b = blockIdx.y, s0 = blockIdx.x * 16;
  const int lane = threadIdx.x & 31, wave = threadIdx.x >> 5;
  const int M = lane & 15, g0 = (lane < 16) ? 0 : 8, n = lane & 15;
  const float L2E = 1.44269504f, SC = 0.125f;   // R^-0.5 / TEMP

  FragBF a0, a1;
  const unsigned short* qr = qv + ((size_t)b * SEQ + s0 + M) * RNK + g0;
  a0.q[0] = *(const v4u*)qr;        a0.q[1] = *(const v4u*)(qr + 16);
  a1.q[0] = *(const v4u*)(qr + 32); a1.q[1] = *(const v4u*)(qr + 48);

  float rm[8], rz[8];
#pragma unroll
  for (int i = 0; i < 8; ++i) { rm[i] = -1e30f; rz[i] = 0.f; }

  for (int lt = wave; lt < SEQ / 16; lt += 4) {
    FragBF b0, b1;
    const unsigned short* kr = kl + ((size_t)b * SEQ + lt * 16 + n) * RNK + g0;
    b0.q[0] = *(const v4u*)kr;        b0.q[1] = *(const v4u*)(kr + 16);
    b1.q[0] = *(const v4u*)(kr + 32); b1.q[1] = *(const v4u*)(kr + 48);
    v8f c = {};
    c = __builtin_amdgcn_wmma_f32_16x16x32_bf16(false, a0.v, false, b0.v, (short)0, c, false, false);
    c = __builtin_amdgcn_wmma_f32_16x16x32_bf16(false, a1.v, false, b1.v, (short)0, c, false, false);
#pragma unroll
    for (int i = 0; i < 8; ++i) {
      float s = c[i] * SC;
      if (s > rm[i]) { rz[i] = rz[i] * exp2f((rm[i] - s) * L2E) + 1.f; rm[i] = s; }
      else           { rz[i] += exp2f((s - rm[i]) * L2E); }
    }
  }
  // combine across the 16 lanes of each half (same 8 rows, different cols)
#pragma unroll
  for (int msk = 1; msk < 16; msk <<= 1) {
#pragma unroll
    for (int i = 0; i < 8; ++i) {
      float om = __shfl_xor(rm[i], msk, 32);
      float oz = __shfl_xor(rz[i], msk, 32);
      float mm = fmaxf(rm[i], om);
      rz[i] = rz[i] * exp2f((rm[i] - mm) * L2E) + oz * exp2f((om - mm) * L2E);
      rm[i] = mm;
    }
  }
  if (lane == 0 || lane == 16) {
    int half = (lane == 0) ? 0 : 8;
#pragma unroll
    for (int i = 0; i < 8; ++i) { lm[wave][half + i] = rm[i]; lz[wave][half + i] = rz[i]; }
  }
  __syncthreads();
  if (threadIdx.x < 16) {
    int row = threadIdx.x;
    float Mv = lm[0][row], Zv = lz[0][row];
#pragma unroll
    for (int w = 1; w < 4; ++w) {
      float om = lm[w][row], oz = lz[w][row];
      float mm = fmaxf(Mv, om);
      Zv = Zv * exp2f((Mv - mm) * L2E) + oz * exp2f((om - mm) * L2E);
      Mv = mm;
    }
    m_ws[(size_t)b * SEQ + s0 + row] = Mv;
    z_ws[(size_t)b * SEQ + s0 + row] = Zv;
  }
}

// ---------------------------------------------------------------------------
// Pass 2: l_imp[b, l] += sum_v exp(s_vl - m_v) / Z_v  (softmax column sums)
// ---------------------------------------------------------------------------
__global__ void colsum_kernel(const unsigned short* __restrict__ qv,
                              const unsigned short* __restrict__ kl,
                              const float* __restrict__ m_ws,
                              const float* __restrict__ z_ws,
                              float* __restrict__ l_imp) {
  const int b = blockIdx.y, s0 = blockIdx.x * 16;
  const int lane = threadIdx.x & 31, wave = threadIdx.x >> 5;
  const int M = lane & 15, g0 = (lane < 16) ? 0 : 8, n = lane & 15;
  const int roff = (lane < 16) ? 0 : 8;
  const float L2E = 1.44269504f, SC = 0.125f;

  FragBF a0, a1;
  const unsigned short* qr = qv + ((size_t)b * SEQ + s0 + M) * RNK + g0;
  a0.q[0] = *(const v4u*)qr;        a0.q[1] = *(const v4u*)(qr + 16);
  a1.q[0] = *(const v4u*)(qr + 32); a1.q[1] = *(const v4u*)(qr + 48);

  float mrow[8], zinv[8];
#pragma unroll
  for (int i = 0; i < 8; ++i) {
    mrow[i] = m_ws[(size_t)b * SEQ + s0 + i + roff];
    zinv[i] = 1.0f / z_ws[(size_t)b * SEQ + s0 + i + roff];
  }
  for (int lt = wave; lt < SEQ / 16; lt += 4) {
    FragBF b0, b1;
    const unsigned short* kr = kl + ((size_t)b * SEQ + lt * 16 + n) * RNK + g0;
    b0.q[0] = *(const v4u*)kr;        b0.q[1] = *(const v4u*)(kr + 16);
    b1.q[0] = *(const v4u*)(kr + 32); b1.q[1] = *(const v4u*)(kr + 48);
    v8f c = {};
    c = __builtin_amdgcn_wmma_f32_16x16x32_bf16(false, a0.v, false, b0.v, (short)0, c, false, false);
    c = __builtin_amdgcn_wmma_f32_16x16x32_bf16(false, a1.v, false, b1.v, (short)0, c, false, false);
    float colsum = 0.f;
#pragma unroll
    for (int i = 0; i < 8; ++i)
      colsum += exp2f((c[i] * SC - mrow[i]) * L2E) * zinv[i];
    colsum += __shfl_xor(colsum, 16, 32);     // merge row-halves (same col)
    if (lane < 16)
      atomicAdd(&l_imp[(size_t)b * SEQ + lt * 16 + lane], colsum);
  }
}

// ---------------------------------------------------------------------------
// top-1024-of-2048 per batch: bitonic sort of packed (score, ~idx) u64 keys
// (ties pick lower index, matching lax.top_k), then sort kept idx ascending.
// ---------------------------------------------------------------------------
__global__ void __launch_bounds__(1024)
topk_kernel(const float* __restrict__ imp, int* __restrict__ idx_out,
            float* __restrict__ idxf_out) {
  __shared__ unsigned long long key[SEQ];
  const int b = blockIdx.x, tid = threadIdx.x;
  for (int e = tid; e < SEQ; e += 1024) {
    float f = imp[(size_t)b * SEQ + e];   // strictly positive -> uint-ordered
    key[e] = ((unsigned long long)__float_as_uint(f) << 32) |
             (unsigned)(SEQ - 1 - e);
  }
  for (unsigned k = 2; k <= SEQ; k <<= 1)
    for (unsigned j = k >> 1; j > 0; j >>= 1) {
      __syncthreads();
      unsigned i = ((tid & ~(j - 1)) << 1) | (tid & (j - 1));
      unsigned l = i | j;
      bool wantDesc = ((i & k) == 0);
      unsigned long long A = key[i], Bv = key[l];
      if ((A < Bv) == wantDesc) { key[i] = Bv; key[l] = A; }
    }
  __syncthreads();
  unsigned myidx = (SEQ - 1) - (unsigned)(key[tid] & 0xFFFFFFFFu);
  __syncthreads();
  unsigned* sidx = (unsigned*)key;
  sidx[tid] = myidx;
  for (unsigned k = 2; k <= NKEEP; k <<= 1)
    for (unsigned j = k >> 1; j > 0; j >>= 1) {
      __syncthreads();
      if (tid < NKEEP / 2) {
        unsigned i = ((tid & ~(j - 1)) << 1) | (tid & (j - 1));
        unsigned l = i | j;
        bool asc = ((i & k) == 0);
        unsigned A = sidx[i], Bv = sidx[l];
        if ((A > Bv) == asc) { sidx[i] = Bv; sidx[l] = A; }
      }
    }
  __syncthreads();
  idx_out[(size_t)b * NKEEP + tid]  = (int)sidx[tid];
  idxf_out[(size_t)b * NKEEP + tid] = (float)sidx[tid];
}

// ---------------------------------------------------------------------------
// Gather (VGPR path): out[b, k, :] = feat[b, idx[b, k], :]
// ---------------------------------------------------------------------------
__global__ void gather_kernel(const float* __restrict__ feat,
                              const int* __restrict__ idx,
                              float* __restrict__ out) {
  const int b = blockIdx.y, k = blockIdx.x, tid = threadIdx.x;
  const int src = idx[(size_t)b * NKEEP + k];
  const float4* s = (const float4*)(feat + ((size_t)b * SEQ + src) * DIM);
  float4*       d = (float4*)(out + ((size_t)b * NKEEP + k) * DIM);
  d[tid] = s[tid];
}

// ---------------------------------------------------------------------------
// Gather (async DMA path): global -> LDS -> global via ASYNCcnt-tracked ops.
// 256 threads x 16 B = one 4 KB row per block; each lane only touches its
// own LDS bytes, so s_wait_asynccnt 0 between load and store is sufficient.
// ---------------------------------------------------------------------------
__global__ void gather_async_kernel(const float* __restrict__ feat,
                                    const int* __restrict__ idx,
                                    float* __restrict__ out) {
  __shared__ float buf[DIM];
  const int b = blockIdx.y, k = blockIdx.x, tid = threadIdx.x;
  const int src = idx[(size_t)b * NKEEP + k];
  const float* sbase = feat + (size_t)b * SEQ * DIM;    // uniform SGPR base
  float*       dbase = out + (size_t)b * NKEEP * DIM;   // uniform SGPR base
  unsigned ldsoff = (unsigned)(size_t)(&buf[0]) + (unsigned)tid * 16u;
  unsigned vin    = (unsigned)src * (DIM * 4u) + (unsigned)tid * 16u;
  unsigned vout   = (unsigned)k   * (DIM * 4u) + (unsigned)tid * 16u;
  asm volatile("global_load_async_to_lds_b128 %0, %1, %2"
               :: "v"(ldsoff), "v"(vin), "s"(sbase) : "memory");
  asm volatile("s_wait_asynccnt 0" ::: "memory");
  asm volatile("global_store_async_from_lds_b128 %0, %1, %2"
               :: "v"(vout), "v"(ldsoff), "s"(dbase) : "memory");
  asm volatile("s_wait_asynccnt 0" ::: "memory");
}

// ---------------------------------------------------------------------------
extern "C" void kernel_launch(void* const* d_in, const int* in_sizes, int n_in,
                              void* d_out, int out_size, void* d_ws, size_t ws_size,
                              hipStream_t stream) {
  const float* vis  = (const float*)d_in[0];   // [16, 2048, 1024]
  const float* lang = (const float*)d_in[1];   // [16, 2048, 1024]
  const float* Wv   = (const float*)d_in[2];   // [64, 1024]
  const float* Wl   = (const float*)d_in[3];   // [64, 1024]
  float* out = (float*)d_out;

  char* ws = (char*)d_ws;
  unsigned short* qv   = (unsigned short*)(ws);                 // 4 MB bf16
  unsigned short* kl   = (unsigned short*)(ws + (size_t)4194304);
  float* m_ws  = (float*)(ws + (size_t)8388608);                // 128 KB each
  float* z_ws  = (float*)(ws + (size_t)8519680);
  float* v_imp = (float*)(ws + (size_t)8650752);
  float* l_imp = (float*)(ws + (size_t)8781824);
  int*   vidx  = (int*)  (ws + (size_t)8912896);
  int*   lidx  = (int*)  (ws + (size_t)8978432);
  unsigned short* Wvbf = (unsigned short*)(ws + (size_t)9043968); // 128 KB
  unsigned short* Wlbf = (unsigned short*)(ws + (size_t)9175040); // 128 KB

  float* out_cv   = out;                          // [16,1024,1024]
  float* out_cl   = out + (size_t)16777216;       // [16,1024,1024]
  float* out_vidx = out + (size_t)33554432;       // [16,1024]
  float* out_lidx = out + (size_t)33571840;       // [16,1024]

  wconv_kernel<<<(RNK * DIM / 2) / 256, 256, 0, stream>>>(Wv, (unsigned*)Wvbf);
  wconv_kernel<<<(RNK * DIM / 2) / 256, 256, 0, stream>>>(Wl, (unsigned*)Wlbf);

  dim3 gTile(SEQ / 16, BN);
  proj_kernel<<<gTile, 128, 0, stream>>>(vis,  Wvbf, qv);
  proj_kernel<<<gTile, 128, 0, stream>>>(lang, Wlbf, kl);

  norm_imp_kernel<<<(BN * SEQ) / 8, 256, 0, stream>>>(vis,  v_imp, 1.0f);
  norm_imp_kernel<<<(BN * SEQ) / 8, 256, 0, stream>>>(lang, l_imp, 0.0f);

  rowstats_kernel<<<gTile, 128, 0, stream>>>(qv, kl, m_ws, z_ws);
  colsum_kernel  <<<gTile, 128, 0, stream>>>(qv, kl, m_ws, z_ws, l_imp);

  topk_kernel<<<BN, 1024, 0, stream>>>(v_imp, vidx, out_vidx);
  topk_kernel<<<BN, 1024, 0, stream>>>(l_imp, lidx, out_lidx);

  gather_kernel      <<<dim3(NKEEP, BN), 256, 0, stream>>>(vis,  vidx, out_cv);
  gather_async_kernel<<<dim3(NKEEP, BN), 256, 0, stream>>>(lang, lidx, out_cl);
}
